// WindowSA_WindowPruningOnly_644245094965
// MI455X (gfx1250) — compile-verified, matrix operand order checked by reading, not run on hardware
//
#include <hip/hip_runtime.h>
#include <hip/hip_bf16.h>
#include <math.h>

// ---------------------------------------------------------------------------
// WindowSA (window pruning) fused transformer block for MI455X (gfx1250).
//  - All GEMMs via v_wmma_f32_16x16x32_bf16 (wave32, fp32 accumulate)
//  - One workgroup (8 waves) per pruned window; everything resident in LDS
//  - Weights transposed+converted to bf16 once (1.5MB, L2-resident)
// ---------------------------------------------------------------------------

#define DIMc 256
#define DHc  32
#define Hc   8
#define Wc   64
#define HIDc 1024
#define EPSc 1e-5f

typedef __attribute__((ext_vector_type(16))) __bf16 v16bf;
typedef __attribute__((ext_vector_type(8)))  __bf16 v8bf;
typedef __attribute__((ext_vector_type(8)))  float  v8f;

// LDS row strides (elements); all chosen so rows are 16-byte aligned.
#define XSTR 264   // bf16 64x256 buffers (region A / region C)
#define QSTR 784   // bf16 64x768 qkv buffer (region B)
#define HSTR 260   // f32  64x256 h buffer (region B reuse)
#define PSTR 72    // bf16 per-wave 16x64 P scratch (region C)

#define REG_A_BYTES (64 * XSTR * 2)                 // 33792
#define REG_B_BYTES (64 * QSTR * 2)                 // 100352 (>= 64*HSTR*4)
#define REG_C_BYTES (64 * XSTR * 2)                 // 33792  (>= 8*16*PSTR*2)
#define SMEM_BYTES  (REG_A_BYTES + REG_B_BYTES + REG_C_BYTES)  // 167936

// Workspace layout (bf16 elements): transposed weights, N-major rows of K.
#define WQKV_OFF 0                       // [768][256]
#define WPRJ_OFF (768 * 256)             // [256][256]
#define WFC1_OFF (WPRJ_OFF + 256 * 256)  // [1024][256]
#define WFC2_OFF (WFC1_OFF + 1024 * 256) // [256][1024]
#define WTOT     (WFC2_OFF + 256 * 1024) // 786432 elements

// ---- WMMA helpers ----------------------------------------------------------

__device__ __forceinline__ v8f wmma_bf16(v16bf a, v16bf b, v8f c) {
  return __builtin_amdgcn_wmma_f32_16x16x32_bf16(false, a, false, b, (short)0, c,
                                                 false, false);
}

// A/B fragment, row-major source: row = M (or N) index for this lane (l&15),
// K covered = k0..k0+31.  Elements 0..7 -> K = k0 + half*8 + e,
// elements 8..15 -> K = k0 + 16 + half*8 + (e-8).  Two contiguous 16B loads.
__device__ __forceinline__ v16bf ld_frag(const __bf16* base, int row, int stride,
                                         int k0, int half) {
  const __bf16* p = base + (size_t)row * stride + k0 + half * 8;
  v8bf lo = *(const v8bf*)(p);
  v8bf hi = *(const v8bf*)(p + 16);
  v16bf f;
#pragma unroll
  for (int i = 0; i < 8; i++) { f[i] = lo[i]; f[i + 8] = hi[i]; }
  return f;
}

// B fragment where the K dimension strides across rows (e.g. V matrix):
// element e reads base[(k0 + kmap(e,half)) * stride + col].
__device__ __forceinline__ v16bf ld_frag_kstrided(const __bf16* base, int col,
                                                  int stride, int k0, int half) {
  v16bf f;
#pragma unroll
  for (int e = 0; e < 16; e++) {
    int k = (e < 8) ? (half * 8 + e) : (16 + half * 8 + (e - 8));
    f[e] = base[(size_t)(k0 + k) * stride + col];
  }
  return f;
}

__device__ __forceinline__ float lane_reduce_add16(float v) {
#pragma unroll
  for (int m = 1; m <= 8; m <<= 1) v += __shfl_xor(v, m, 32);
  return v;
}
__device__ __forceinline__ float lane_reduce_max16(float v) {
#pragma unroll
  for (int m = 1; m <= 8; m <<= 1) v = fmaxf(v, __shfl_xor(v, m, 32));
  return v;
}
__device__ __forceinline__ float lane_reduce_add32(float v) {
#pragma unroll
  for (int m = 1; m <= 16; m <<= 1) v += __shfl_xor(v, m, 32);
  return v;
}

// ---- Kernel 0: transpose-convert weights to bf16 ---------------------------

__global__ void wconv_kernel(const float* __restrict__ qkv_w,
                             const float* __restrict__ proj_w,
                             const float* __restrict__ fc1_w,
                             const float* __restrict__ fc2_w,
                             __bf16* __restrict__ ws) {
  int i = blockIdx.x * 256 + threadIdx.x;
  if (i >= WTOT) return;
  if (i < WPRJ_OFF) {                       // qkvT[n][k] = qkv_w[k][n]
    int j = i, n = j / 256, k = j % 256;
    ws[i] = (__bf16)qkv_w[k * 768 + n];
  } else if (i < WFC1_OFF) {                // projT[n][k] = proj_w[k][n]
    int j = i - WPRJ_OFF, n = j / 256, k = j % 256;
    ws[i] = (__bf16)proj_w[k * 256 + n];
  } else if (i < WFC2_OFF) {                // fc1T[n][k] = fc1_w[k][n]
    int j = i - WFC1_OFF, n = j / 256, k = j % 256;
    ws[i] = (__bf16)fc1_w[k * 1024 + n];
  } else {                                  // fc2T[n][k] = fc2_w[k][n]
    int j = i - WFC2_OFF, n = j / 1024, k = j % 1024;
    ws[i] = (__bf16)fc2_w[k * 256 + n];
  }
}

// ---- Kernel 1: LayerNorm of every row of x -> d_out (this is X) ------------

__global__ void ln_kernel(const float* __restrict__ x,
                          const float* __restrict__ g,
                          const float* __restrict__ b,
                          float* __restrict__ out, int nrows) {
  int row = blockIdx.x * 8 + (threadIdx.x >> 5);
  if (row >= nrows) return;
  int lane = threadIdx.x & 31;
  const float* xr = x + (size_t)row * DIMc;
  float v[8];
  float s = 0.f;
#pragma unroll
  for (int j = 0; j < 8; j++) { v[j] = xr[lane + 32 * j]; s += v[j]; }
  s = lane_reduce_add32(s);
  float mu = s * (1.0f / DIMc);
  float var = 0.f;
#pragma unroll
  for (int j = 0; j < 8; j++) { float d = v[j] - mu; var += d * d; }
  var = lane_reduce_add32(var) * (1.0f / DIMc);
  float inv = rsqrtf(var + EPSc);
  float* o = out + (size_t)row * DIMc;
#pragma unroll
  for (int j = 0; j < 8; j++) {
    int c = lane + 32 * j;
    o[c] = (v[j] - mu) * inv * g[c] + b[c];
  }
}

// ---- Kernel 2: fused per-window transformer block --------------------------

__global__ __launch_bounds__(256, 1) void fused_block(
    float* __restrict__ Out,            // d_out: X, rows of selected windows overwritten
    const int* __restrict__ index,
    const float* __restrict__ qkv_b, const float* __restrict__ proj_b,
    const float* __restrict__ n2g, const float* __restrict__ n2b,
    const float* __restrict__ fc1_b, const float* __restrict__ fc2_b,
    const __bf16* __restrict__ wsw) {
  extern __shared__ char smem[];
  __bf16* sA   = (__bf16*)smem;                       // xs -> attn_out -> hn (bf16 64xXSTR)
  char*   regB = smem + REG_A_BYTES;
  __bf16* sQKV = (__bf16*)regB;                       // 64x768 bf16
  float*  sH   = (float*)regB;                        // 64x256 f32 (reuse after attn)
  char*   regC = regB + REG_B_BYTES;
  __bf16* sP   = (__bf16*)regC;                       // per-wave 16x64 P scratch
  __bf16* sG   = (__bf16*)regC;                       // GELU chunk 64x256 (reuse)

  const int tid  = threadIdx.x;
  const int wave = tid >> 5;
  const int lane = tid & 31;
  const int ln   = lane & 15;
  const int half = lane >> 4;
  const int win  = index[blockIdx.x];
  const float* xwin = Out + (size_t)win * (Wc * DIMc);   // X rows (read-only until end)

  const __bf16* qkvT = wsw + WQKV_OFF;
  const __bf16* prjT = wsw + WPRJ_OFF;
  const __bf16* fc1T = wsw + WFC1_OFF;
  const __bf16* fc2T = wsw + WFC2_OFF;

  // ---- Phase 0: stage xs (= X[win]) into LDS as bf16 -----------------------
  for (int i = tid; i < (Wc * DIMc) / 4; i += 256) {
    int r = (i * 4) / DIMc, c = (i * 4) % DIMc;
    float4 v = *(const float4*)(xwin + (size_t)r * DIMc + c);
    __bf16* d = sA + r * XSTR + c;
    d[0] = (__bf16)v.x; d[1] = (__bf16)v.y; d[2] = (__bf16)v.z; d[3] = (__bf16)v.w;
  }
  __syncthreads();

  // ---- Phase 1: QKV = xs @ qkv_w + qkv_b  (64x768, bf16 out) ---------------
  for (int t = wave; t < 4 * 48; t += 8) {
    int m0 = (t / 48) * 16, n0 = (t % 48) * 16;
    v8f acc = {};
#pragma unroll
    for (int kk = 0; kk < DIMc; kk += 32) {
      v16bf a = ld_frag(sA,   m0 + ln, XSTR, kk, half);
      v16bf b = ld_frag(qkvT, n0 + ln, 256,  kk, half);
      acc = wmma_bf16(a, b, acc);
    }
    int col = n0 + ln;
    float bias = qkv_b[col];
#pragma unroll
    for (int r = 0; r < 8; r++)
      sQKV[(m0 + r + 8 * half) * QSTR + col] = (__bf16)(acc[r] + bias);
  }
  __syncthreads();

  // ---- Phase 2: attention (per 16-row strip per head) ----------------------
  const float scale = 0.17677669529663687f;   // 1/sqrt(32)
  __bf16* sPw = sP + wave * 16 * PSTR;
  for (int s = wave; s < Hc * (Wc / 16); s += 8) {
    int head = s >> 2, m0 = (s & 3) * 16;
    int qc = head * 96, kc = qc + 32, vc = qc + 64;

    // S = (Q Kt) * scale : 16x64, four 16x16 tiles, single K=32 WMMA each
    v8f sacc[4];
#pragma unroll
    for (int nt = 0; nt < 4; nt++) {
      v8f z = {};
      v16bf a = ld_frag(sQKV + qc, m0 + ln,        QSTR, 0, half);
      v16bf b = ld_frag(sQKV + kc, nt * 16 + ln,   QSTR, 0, half);
      sacc[nt] = wmma_bf16(a, b, z);
    }
    // softmax over 64 cols; row = m0 + r + 8*half lives in lanes sharing `half`
#pragma unroll
    for (int r = 0; r < 8; r++) {
      float mx = -3.0e38f;
#pragma unroll
      for (int nt = 0; nt < 4; nt++) mx = fmaxf(mx, sacc[nt][r] * scale);
      mx = lane_reduce_max16(mx);
      float sum = 0.f;
#pragma unroll
      for (int nt = 0; nt < 4; nt++) {
        float e = __expf(sacc[nt][r] * scale - mx);
        sacc[nt][r] = e;
        sum += e;
      }
      sum = lane_reduce_add16(sum);
      float inv = 1.0f / sum;
#pragma unroll
      for (int nt = 0; nt < 4; nt++) sacc[nt][r] *= inv;
    }
    // stash P (bf16) in per-wave scratch so it can be re-read in A-layout
#pragma unroll
    for (int nt = 0; nt < 4; nt++)
#pragma unroll
      for (int r = 0; r < 8; r++)
        sPw[(r + 8 * half) * PSTR + nt * 16 + ln] = (__bf16)sacc[nt][r];

    // out = P @ V : 16x32 (two tiles), K = 64
#pragma unroll
    for (int nt = 0; nt < 2; nt++) {
      v8f oacc = {};
#pragma unroll
      for (int kk = 0; kk < 64; kk += 32) {
        v16bf a = ld_frag(sPw, ln, PSTR, kk, half);
        v16bf b = ld_frag_kstrided(sQKV + vc + nt * 16, ln, QSTR, kk, half);
        oacc = wmma_bf16(a, b, oacc);
      }
#pragma unroll
      for (int r = 0; r < 8; r++)
        sA[(m0 + r + 8 * half) * XSTR + head * DHc + nt * 16 + ln] = (__bf16)oacc[r];
    }
  }
  __syncthreads();

  // ---- Phase 3: h = attn @ proj_w + proj_b + shortcut ----------------------
  for (int t = wave; t < 64; t += 8) {
    int m0 = (t / 16) * 16, n0 = (t % 16) * 16;
    v8f acc = {};
#pragma unroll
    for (int kk = 0; kk < DIMc; kk += 32) {
      v16bf a = ld_frag(sA,   m0 + ln, XSTR, kk, half);
      v16bf b = ld_frag(prjT, n0 + ln, 256,  kk, half);
      acc = wmma_bf16(a, b, acc);
    }
    int col = n0 + ln;
    float pb = proj_b[col];
#pragma unroll
    for (int r = 0; r < 8; r++) {
      int row = m0 + r + 8 * half;
      sH[row * HSTR + col] = acc[r] + pb + xwin[(size_t)row * DIMc + col];
    }
  }
  __syncthreads();

  // ---- Phase 4: LN2(h) -> hn (bf16, region A) ------------------------------
  for (int row = wave; row < Wc; row += 8) {
    float v[8];
    float s = 0.f;
#pragma unroll
    for (int j = 0; j < 8; j++) { v[j] = sH[row * HSTR + lane + 32 * j]; s += v[j]; }
    s = lane_reduce_add32(s);
    float mu = s * (1.0f / DIMc);
    float var = 0.f;
#pragma unroll
    for (int j = 0; j < 8; j++) { float d = v[j] - mu; var += d * d; }
    var = lane_reduce_add32(var) * (1.0f / DIMc);
    float inv = rsqrtf(var + EPSc);
#pragma unroll
    for (int j = 0; j < 8; j++) {
      int c = lane + 32 * j;
      sA[row * XSTR + c] = (__bf16)((v[j] - mu) * inv * n2g[c] + n2b[c]);
    }
  }

  // ---- Phase 5: MLP, HID chunked by 256; fc2 accumulates in registers ------
  v8f h2acc[8];
#pragma unroll
  for (int i = 0; i < 8; i++) h2acc[i] = (v8f){};
  for (int cc = 0; cc < HIDc / 256; cc++) {
    __syncthreads();   // hn ready (first iter) / previous G fully consumed
    // G = gelu(hn @ fc1_w[:, cc*256:+256] + b1) -> sG bf16
    for (int t = wave; t < 64; t += 8) {
      int m0 = (t / 16) * 16, n0 = (t % 16) * 16;
      v8f acc = {};
#pragma unroll
      for (int kk = 0; kk < DIMc; kk += 32) {
        v16bf a = ld_frag(sA,   m0 + ln,               XSTR, kk, half);
        v16bf b = ld_frag(fc1T, cc * 256 + n0 + ln,    256,  kk, half);
        acc = wmma_bf16(a, b, acc);
      }
      int col = n0 + ln;
      float b1 = fc1_b[cc * 256 + col];
#pragma unroll
      for (int r = 0; r < 8; r++) {
        float xv = acc[r] + b1;
        float g = 0.5f * xv * (1.0f + erff(xv * 0.70710678118654752f));
        sG[(m0 + r + 8 * half) * XSTR + col] = (__bf16)g;
      }
    }
    __syncthreads();
    // h2 += G @ fc2_w[cc*256:+256, :]
#pragma unroll
    for (int i = 0; i < 8; i++) {
      int t = wave + 8 * i;
      int m0 = (t / 16) * 16, n0 = (t % 16) * 16;
#pragma unroll
      for (int kk = 0; kk < 256; kk += 32) {
        v16bf a = ld_frag(sG,   m0 + ln, XSTR, kk, half);
        v16bf b = ld_frag(fc2T, n0 + ln, HIDc, cc * 256 + kk, half);
        h2acc[i] = wmma_bf16(a, b, h2acc[i]);
      }
    }
  }

  // ---- Phase 6: write h + h2 + fc2_b into d_out rows -----------------------
  float* owin = Out + (size_t)win * (Wc * DIMc);
#pragma unroll
  for (int i = 0; i < 8; i++) {
    int t = wave + 8 * i;
    int m0 = (t / 16) * 16, n0 = (t % 16) * 16;
    int col = n0 + ln;
    float b2 = fc2_b[col];
#pragma unroll
    for (int r = 0; r < 8; r++) {
      int row = m0 + r + 8 * half;
      owin[(size_t)row * DIMc + col] = sH[row * HSTR + col] + h2acc[i][r] + b2;
    }
  }
}

// ---------------------------------------------------------------------------

extern "C" void kernel_launch(void* const* d_in, const int* in_sizes, int n_in,
                              void* d_out, int out_size, void* d_ws, size_t ws_size,
                              hipStream_t stream) {
  const float* x       = (const float*)d_in[0];
  const int*   index   = (const int*)d_in[1];
  const float* n1g     = (const float*)d_in[3];
  const float* n1b     = (const float*)d_in[4];
  const float* qkv_w   = (const float*)d_in[5];
  const float* qkv_b   = (const float*)d_in[6];
  const float* proj_w  = (const float*)d_in[7];
  const float* proj_b  = (const float*)d_in[8];
  const float* n2g     = (const float*)d_in[9];
  const float* n2b     = (const float*)d_in[10];
  const float* fc1_w   = (const float*)d_in[11];
  const float* fc1_b   = (const float*)d_in[12];
  const float* fc2_w   = (const float*)d_in[13];
  const float* fc2_b   = (const float*)d_in[14];
  float*  out = (float*)d_out;
  __bf16* wsw = (__bf16*)d_ws;

  const int M     = in_sizes[1];             // 2048 selected windows
  const int nrows = in_sizes[0] / DIMc;      // 4096*64 rows

  (void)n_in; (void)out_size; (void)ws_size;

  hipFuncSetAttribute((const void*)fused_block,
                      hipFuncAttributeMaxDynamicSharedMemorySize, SMEM_BYTES);

  wconv_kernel<<<(WTOT + 255) / 256, 256, 0, stream>>>(qkv_w, proj_w, fc1_w, fc2_w, wsw);
  ln_kernel<<<(nrows + 7) / 8, 256, 0, stream>>>(x, n1g, n1b, out, nrows);
  fused_block<<<M, 256, SMEM_BYTES, stream>>>(out, index, qkv_b, proj_b,
                                              n2g, n2b, fc1_b, fc2_b, wsw);
}
